// EMSA_6863357739038
// MI455X (gfx1250) — matrix operand order repaired and checked by reading, not compiled, not run on hardware
//
#include <hip/hip_runtime.h>
#include <hip/hip_bf16.h>

// ---------------------------------------------------------------------------
// EMSA (PVT-style spatial-reduction attention) for MI455X / gfx1250.
// All GEMM-like work via v_wmma_f32_16x16x32_bf16 (f32 accumulate).
// B=32, N=1024, DIM=512, HEAD=8, d=64, kv tokens m=256, scale=DIM^-0.5.
// ---------------------------------------------------------------------------

typedef __attribute__((ext_vector_type(16))) __bf16 v16bf;
typedef __attribute__((ext_vector_type(8)))  __bf16 bf16x8;
typedef __attribute__((ext_vector_type(8)))  float  v8f;

#define DEV static __device__ __forceinline__

#define BATCH 32
#define NQ    1024
#define DIMC  512
#define HEADS 8
#define HDIM  64
#define MKV   256
#define SCALE 0.04419417382415922f  // 512^-0.5

DEV __bf16 f2bf(float f) {
  unsigned int u = __float_as_uint(f);
  u += 0x7FFFu + ((u >> 16) & 1u);           // round-to-nearest-even
  unsigned short s = (unsigned short)(u >> 16);
  __bf16 r;
  __builtin_memcpy(&r, &s, sizeof(r));
  return r;
}

DEV v8f wmma_bf16(v16bf a, v16bf b, v8f c) {
  // v_wmma_f32_16x16x32_bf16: D = A(16x32) * B(32x16) + C(16x16 f32)
  return __builtin_amdgcn_wmma_f32_16x16x32_bf16(
      false, a, false, b, (short)0, c, false, false);
}

// A-fragment per ISA layout: lane half h -> elems {kb+8h+0..7, kb+16+8h+0..7}.
// Caller passes p = row_base + kb + 8*half; we read p[0..7] and p[16..23].
DEV v16bf load_afrag(const __bf16* p) {
  bf16x8 lo = *reinterpret_cast<const bf16x8*>(p);
  bf16x8 hi = *reinterpret_cast<const bf16x8*>(p + 16);
  v16bf r;
#pragma unroll
  for (int i = 0; i < 8; ++i) { r[i] = lo[i]; r[i + 8] = hi[i]; }
  return r;
}

// B-fragment: same K order as A (kb+8h+0..7, kb+16+8h+0..7); p = Bt_row + kb + 16*half.
DEV v16bf load_bfrag(const __bf16* p) {
  return load_afrag(p - 8 * ((threadIdx.x & 31) >> 4));
}

// ---------------------------------------------------------------------------
// Casting / packing kernels
// ---------------------------------------------------------------------------
__global__ void cast_f32_bf16(const float* __restrict__ in,
                              __bf16* __restrict__ out, int n) {
  int i = blockIdx.x * blockDim.x + threadIdx.x;
  if (i < n) out[i] = f2bf(in[i]);
}

// conv_w [O=1024][C=512][3][3] f32 -> wpack [tap=9][O][C] bf16 (Bt rows per tap)
__global__ void pack_conv_w(const float* __restrict__ w, __bf16* __restrict__ out) {
  int i = blockIdx.x * blockDim.x + threadIdx.x;
  if (i >= 1024 * 512 * 9) return;
  int tap = i % 9;
  int c   = (i / 9) % 512;
  int o   = i / (9 * 512);
  out[((size_t)tap * 1024 + o) * 512 + c] = f2bf(w[i]);
}

// ---------------------------------------------------------------------------
// Conv 3x3 s2 p1 (512 -> 1024) as implicit GEMM over 9 taps.
// One wave per 16-pixel x 64-outchannel tile. Grid: 32 b * 16 i * 16 nt = 8192 waves.
// Writes convout [b*256 + i*16 + j][o]  (i.e. [B, m, 2*DIM] row-major, f32).
// ---------------------------------------------------------------------------
__global__ void conv_gemm(const __bf16* __restrict__ xin,   // [B,1024,512]
                          const __bf16* __restrict__ wpack, // [9,1024,512]
                          const float*  __restrict__ bias,  // [1024]
                          float* __restrict__ out) {        // [B*256,1024]
  int wid  = (blockIdx.x * blockDim.x + threadIdx.x) >> 5;
  int nt   = wid & 15;          // 64-wide out-channel block
  int i    = (wid >> 4) & 15;   // output row (first spatial dim)
  int b    = wid >> 8;
  int lane = threadIdx.x & 31;
  int half = lane >> 4, sub = lane & 15;
  int j = sub;                  // output col (second spatial dim) = A-row

  v8f acc[4] = {{}, {}, {}, {}};

#pragma unroll
  for (int tap = 0; tap < 9; ++tap) {
    int di = tap / 3, dj = tap % 3;
    int y = 2 * i - 1 + di;
    int x = 2 * j - 1 + dj;
    bool valid = (y >= 0) & (y < 32) & (x >= 0) & (x < 32);
    const __bf16* wt = wpack + (size_t)tap * (1024 * 512);
    for (int ks = 0; ks < 16; ++ks) {
      int kb = ks * 32;
      v16bf af = {};
      if (valid) {
        const __bf16* arow =
            xin + ((size_t)b * 1024 + (size_t)(y * 32 + x)) * 512;
        af = load_afrag(arow + kb + 8 * half);
      }
#pragma unroll
      for (int t = 0; t < 4; ++t) {
        const __bf16* br = wt + (size_t)(nt * 64 + t * 16 + sub) * 512;
        acc[t] = wmma_bf16(af, load_bfrag(br + kb + 16 * half), acc[t]);
      }
    }
  }

  int rowg = b * 256 + i * 16;  // + (v + 8*half)
#pragma unroll
  for (int t = 0; t < 4; ++t) {
    int col = nt * 64 + t * 16 + sub;
    float bv = bias[col];
#pragma unroll
    for (int v = 0; v < 8; ++v) {
      out[(size_t)(rowg + v + 8 * half) * 1024 + col] = acc[t][v] + bv;
    }
  }
}

// ---------------------------------------------------------------------------
// LayerNorm over 1024 channels per (b,m) row, then split:
//   k  -> [B, m, 512] bf16 (Bt layout for QK^T)
//   vT -> [B, H, 64, 256] bf16 (Bt layout for att*V)
// One wave per row; 8192 rows.
// ---------------------------------------------------------------------------
__global__ void ln_split(const float* __restrict__ conv,  // [8192,1024]
                         const float* __restrict__ gamma,
                         const float* __restrict__ beta,
                         __bf16* __restrict__ kout,       // [B,256,512]
                         __bf16* __restrict__ vt) {       // [B,8,64,256]
  int row  = (blockIdx.x * blockDim.x + threadIdx.x) >> 5;
  int lane = threadIdx.x & 31;
  const float* r = conv + (size_t)row * 1024;

  float vals[32];
  float s = 0.f, s2 = 0.f;
#pragma unroll
  for (int i = 0; i < 32; ++i) {
    float f = r[lane + 32 * i];
    vals[i] = f; s += f; s2 += f * f;
  }
#pragma unroll
  for (int m = 1; m < 32; m <<= 1) {
    s  += __shfl_xor(s, m, 32);
    s2 += __shfl_xor(s2, m, 32);
  }
  float mean = s * (1.0f / 1024.0f);
  float var  = s2 * (1.0f / 1024.0f) - mean * mean;
  float rstd = rsqrtf(var + 1e-5f);

  int b = row >> 8, m = row & 255;
#pragma unroll
  for (int i = 0; i < 32; ++i) {
    int col = lane + 32 * i;
    float nv = (vals[i] - mean) * rstd * gamma[col] + beta[col];
    __bf16 bv = f2bf(nv);
    if (col < 512) {
      kout[(size_t)row * 512 + col] = bv;
    } else {
      int c = col - 512, h = c >> 6, dd = c & 63;
      vt[(((size_t)(b * 8 + h)) * 64 + dd) * 256 + m] = bv;
    }
  }
}

// ---------------------------------------------------------------------------
// C[M,512] = A[M,512](bf16) * Bt^T,  Bt [512,512] bf16 row-major (= W row-major).
// Wave tile 16x64; 16 K-steps. BF16_OUT selects bf16 vs f32 output.
// ---------------------------------------------------------------------------
template <bool BF16_OUT>
__global__ void gemm_nt(const __bf16* __restrict__ A,
                        const __bf16* __restrict__ Bt,
                        float* __restrict__ outf,
                        __bf16* __restrict__ outb) {
  int wid  = (blockIdx.x * blockDim.x + threadIdx.x) >> 5;
  int nt   = wid & 7;   // 8 x 64-wide col blocks
  int mt   = wid >> 3;  // 16-row blocks (M = 32768)
  int lane = threadIdx.x & 31;
  int half = lane >> 4, sub = lane & 15;

  const __bf16* arow = A + (size_t)(mt * 16 + sub) * 512;
  v8f acc[4] = {{}, {}, {}, {}};
  for (int ks = 0; ks < 16; ++ks) {
    int kb = ks * 32;
    v16bf af = load_afrag(arow + kb + 8 * half);
#pragma unroll
    for (int t = 0; t < 4; ++t) {
      const __bf16* br = Bt + (size_t)(nt * 64 + t * 16 + sub) * 512;
      acc[t] = wmma_bf16(af, load_bfrag(br + kb + 16 * half), acc[t]);
    }
  }
#pragma unroll
  for (int t = 0; t < 4; ++t) {
    int col = nt * 64 + t * 16 + sub;
#pragma unroll
    for (int v = 0; v < 8; ++v) {
      size_t idx = (size_t)(mt * 16 + v + 8 * half) * 512 + col;
      if (BF16_OUT) outb[idx] = f2bf(acc[t][v]);
      else          outf[idx] = acc[t][v];
    }
  }
}

// ---------------------------------------------------------------------------
// Fused attention per (b, h, 16-query tile):
//   alpha = scale * q k^T  (16x256, 32 WMMA), softmax rows (shfl-xor over 16
//   lanes), transpose att through per-wave LDS (C-layout -> A-layout),
//   out = att * v (32 WMMA), store bf16 [B,1024,512].
// 16384 waves, 128 threads/block (4 waves, 32KB LDS).
// ---------------------------------------------------------------------------
__global__ void attention(const __bf16* __restrict__ q,   // [B,1024,512]
                          const __bf16* __restrict__ k,   // [B,256,512]
                          const __bf16* __restrict__ vt,  // [B,8,64,256]
                          __bf16* __restrict__ out) {     // [B,1024,512]
  __shared__ __bf16 smem[4][16 * 256];
  int wid  = (blockIdx.x * blockDim.x + threadIdx.x) >> 5;
  int qt   = wid & 63;
  int h    = (wid >> 6) & 7;
  int b    = wid >> 9;
  int lane = threadIdx.x & 31;
  int half = lane >> 4, sub = lane & 15;
  __bf16* satt = &smem[(threadIdx.x >> 5) & 3][0];

  // --- alpha = q k^T (K = 64, two WMMA steps per 16-col tile) ---
  const __bf16* qrow =
      q + ((size_t)b * 1024 + (size_t)(qt * 16 + sub)) * 512 + h * 64;
  v16bf qa0 = load_afrag(qrow + 0  + 8 * half);
  v16bf qa1 = load_afrag(qrow + 32 + 8 * half);

  v8f a[16];
#pragma unroll
  for (int t = 0; t < 16; ++t) {
    const __bf16* krow =
        k + ((size_t)b * 256 + (size_t)(t * 16 + sub)) * 512 + h * 64;
    v8f c = {};
    c = wmma_bf16(qa0, load_bfrag(krow + 0  + 16 * half), c);
    c = wmma_bf16(qa1, load_bfrag(krow + 32 + 16 * half), c);
    a[t] = c;
  }

  // --- scale + row softmax (row = v + 8*half, cols striped across 16 lanes) ---
#pragma unroll
  for (int v = 0; v < 8; ++v) {
    float mx = -3.0e38f;
#pragma unroll
    for (int t = 0; t < 16; ++t) {
      float x = a[t][v] * SCALE;
      a[t][v] = x;
      mx = fmaxf(mx, x);
    }
    mx = fmaxf(mx, __shfl_xor(mx, 1, 32));
    mx = fmaxf(mx, __shfl_xor(mx, 2, 32));
    mx = fmaxf(mx, __shfl_xor(mx, 4, 32));
    mx = fmaxf(mx, __shfl_xor(mx, 8, 32));
    float s = 0.f;
#pragma unroll
    for (int t = 0; t < 16; ++t) {
      float e = __expf(a[t][v] - mx);
      a[t][v] = e;
      s += e;
    }
    s += __shfl_xor(s, 1, 32);
    s += __shfl_xor(s, 2, 32);
    s += __shfl_xor(s, 4, 32);
    s += __shfl_xor(s, 8, 32);
    float inv = 1.0f / s;
#pragma unroll
    for (int t = 0; t < 16; ++t) a[t][v] *= inv;
  }

  // --- transpose att through LDS: C-layout -> row-major bf16 [16][256] ---
#pragma unroll
  for (int t = 0; t < 16; ++t) {
#pragma unroll
    for (int v = 0; v < 8; ++v) {
      satt[(v + 8 * half) * 256 + t * 16 + sub] = f2bf(a[t][v]);
    }
  }
  asm volatile("s_wait_dscnt 0x0" ::: "memory");  // same-wave LDS RAW fence

  // --- out = att * v  (K = 256, 8 steps; 4 x 16-wide d tiles) ---
  v8f oacc[4] = {{}, {}, {}, {}};
  const __bf16* vbase = vt + (((size_t)(b * 8 + h)) * 64) * 256;
  for (int ks = 0; ks < 8; ++ks) {
    int kb = ks * 32;
    v16bf af = load_afrag(&satt[sub * 256 + kb + 8 * half]);
#pragma unroll
    for (int t = 0; t < 4; ++t) {
      const __bf16* vr = vbase + (size_t)(t * 16 + sub) * 256;
      oacc[t] = wmma_bf16(af, load_bfrag(vr + kb + 16 * half), oacc[t]);
    }
  }
#pragma unroll
  for (int t = 0; t < 4; ++t) {
#pragma unroll
    for (int v = 0; v < 8; ++v) {
      size_t idx = ((size_t)b * 1024 + (size_t)(qt * 16 + v + 8 * half)) * 512 +
                   h * 64 + t * 16 + sub;
      out[idx] = f2bf(oacc[t][v]);
    }
  }
}

// ---------------------------------------------------------------------------
// Host launcher
// ---------------------------------------------------------------------------
extern "C" void kernel_launch(void* const* d_in, const int* in_sizes, int n_in,
                              void* d_out, int out_size, void* d_ws, size_t ws_size,
                              hipStream_t stream) {
  const float* feats  = (const float*)d_in[0];
  const float* W_q    = (const float*)d_in[1];
  const float* conv_w = (const float*)d_in[2];
  const float* conv_b = (const float*)d_in[3];
  const float* ln_g   = (const float*)d_in[4];
  const float* ln_b   = (const float*)d_in[5];
  const float* W_out  = (const float*)d_in[6];
  float* out = (float*)d_out;

  char* ws = (char*)d_ws;
  size_t off = 0;
  auto take = [&](size_t bytes) -> char* {
    char* p = ws + off;
    off += (bytes + 255) & ~(size_t)255;
    return p;
  };

  const size_t n_feats = (size_t)BATCH * NQ * DIMC;          // 16777216
  __bf16* feats_bf = (__bf16*)take(n_feats * 2);             // 32 MB
  __bf16* q_bf     = (__bf16*)take(n_feats * 2);             // 32 MB
  float*  convout  = (float*)take((size_t)BATCH * MKV * 1024 * 4);  // 32 MB
  __bf16* k_bf     = (__bf16*)take((size_t)BATCH * MKV * DIMC * 2); // 8 MB
  __bf16* vt_bf    = (__bf16*)take((size_t)BATCH * MKV * DIMC * 2); // 8 MB
  __bf16* wq_bf    = (__bf16*)take((size_t)DIMC * DIMC * 2);
  __bf16* wout_bf  = (__bf16*)take((size_t)DIMC * DIMC * 2);
  __bf16* wpack    = (__bf16*)take((size_t)9 * 1024 * 512 * 2);     // 9.4 MB
  // attention output (bf16, 32 MB) reuses the conv-output buffer: conv output
  // is fully consumed by ln_split before attention runs (same stream, in order).
  __bf16* attn_bf  = (__bf16*)convout;

  // 1) casts / packing
  cast_f32_bf16<<<(int)(n_feats / 256), 256, 0, stream>>>(feats, feats_bf,
                                                          (int)n_feats);
  cast_f32_bf16<<<1024, 256, 0, stream>>>(W_q,   wq_bf,   DIMC * DIMC);
  cast_f32_bf16<<<1024, 256, 0, stream>>>(W_out, wout_bf, DIMC * DIMC);
  pack_conv_w<<<(9 * 1024 * 512 + 255) / 256, 256, 0, stream>>>(conv_w, wpack);

  // 2) conv (implicit GEMM, 8192 waves)
  conv_gemm<<<1024, 256, 0, stream>>>(feats_bf, wpack, conv_b, convout);

  // 3) LayerNorm + k/v split (8192 waves)
  ln_split<<<1024, 256, 0, stream>>>(convout, ln_g, ln_b, k_bf, vt_bf);

  // 4) q = feats @ W_q^T   (16384 waves)
  gemm_nt<true><<<2048, 256, 0, stream>>>(feats_bf, wq_bf, nullptr, q_bf);

  // 5) fused attention (16384 waves)
  attention<<<4096, 128, 0, stream>>>(q_bf, k_bf, vt_bf, attn_bf);

  // 6) out = attn @ W_out^T (f32 output)
  gemm_nt<false><<<2048, 256, 0, stream>>>(attn_bf, wout_bf, out, nullptr);
}
